// FastMKAAttention_5875515260950
// MI455X (gfx1250) — matrix operand order repaired and checked by reading, not compiled, not run on hardware
//
#include <hip/hip_runtime.h>
#include <hip/hip_bf16.h>
#include <math.h>

// ---------------------------------------------------------------------------
// MI455X (gfx1250, wave32) FastMKAAttention.
// All matmuls on v_wmma_f32_16x16x32_f16 (fp32 accum). Weights/activations
// pre-converted to f16 once (halves dominant streaming traffic, deletes all
// cvt from GEMM hot loops). A tiles staged to LDS with
// GLOBAL_LOAD_ASYNC_TO_LDS_B128 (ASYNCcnt) and shared by 4 waves/block.
// ---------------------------------------------------------------------------

typedef _Float16 f16;
typedef __attribute__((ext_vector_type(16))) _Float16 v16h;
typedef __attribute__((ext_vector_type(8)))  float    v8f;

#define WMMA16(a, b, c) \
  __builtin_amdgcn_wmma_f32_16x16x32_f16(false, (a), false, (b), (short)0, (c), false, false)

#define AS1 __attribute__((address_space(1)))
#define AS3 __attribute__((address_space(3)))

// exact pointee type of the async-LDS builtin params (from hipcc diagnostic)
typedef int v4i_vec __attribute__((vector_size(16)));

#if __has_builtin(__builtin_amdgcn_global_load_async_to_lds_b128) && \
    __has_builtin(__builtin_amdgcn_s_wait_asynccnt)
#define HAVE_ASYNC_LDS 1
#else
#define HAVE_ASYNC_LDS 0
#endif

// Problem constants
#define BB 2
#define TT 1024
#define DD 2048
#define HH 16
#define DHD 128
#define HDN 1024   /* d/2 */

// --------------------------- fragment loaders ------------------------------
// A-matrix 16x32 f16 layout: lane L: m = L%16, half = L/16.
// elems [0..15] = K 16*half .. 16*half+15 (contiguous)
__device__ __forceinline__ v16h load_a_frag_lds16(const f16* lds_a, int lane) {
  int m = lane & 15, half = lane >> 4;
  const f16* ap = lds_a + m * 32 + 16 * half;
  v16h a;
#pragma unroll
  for (int e = 0; e < 16; ++e) a[e] = ap[e];
  return a;
}

// Same A layout from global f16 [., lda]
__device__ __forceinline__ v16h load_a_frag_g16(const f16* __restrict__ A, int lda,
                                                int row0, int k0, int lane) {
  int m = lane & 15, half = lane >> 4;
  const f16* ap = A + (size_t)(row0 + m) * lda + k0 + 16 * half;
  v16h a;
#pragma unroll
  for (int e = 0; e < 16; ++e) a[e] = ap[e];
  return a;
}

// B-matrix 32x16 f16 (KxN) from row-major f16 weight W16[N,K]:
// lane: n = L%16, half = L/16, elem e = K (e + 16*half) -> 16 contiguous f16.
__device__ __forceinline__ v16h load_b_frag_w16(const f16* __restrict__ W, int ldw,
                                                int n0, int k0, int lane) {
  int n = lane & 15, half = lane >> 4;
  const f16* wp = W + (size_t)(n0 + n) * ldw + k0 + 16 * half;
  v16h b;
#pragma unroll
  for (int e = 0; e < 16; ++e) b[e] = wp[e];
  return b;
}

// B fragment from f16 K-cache [t, dh]: scores B[kdim=dd][n=token]
__device__ __forceinline__ v16h load_b_frag_k(const f16* __restrict__ kf,
                                              int tok0, int kb, int lane) {
  int n = lane & 15, half = lane >> 4;
  const f16* p = kf + (size_t)(tok0 + n) * DHD + kb + 16 * half;
  v16h b;
#pragma unroll
  for (int e = 0; e < 16; ++e) b[e] = p[e];
  return b;
}

// B fragment from f16 V-cache [t, dh]: PV B[kdim=token][n=dd]
__device__ __forceinline__ v16h load_b_frag_v(const f16* __restrict__ vf,
                                              int tok0, int n0, int lane) {
  int n = lane & 15, half = lane >> 4;
  const f16* p = vf + (size_t)(tok0 + 16 * half) * DHD + n0 + n;
  v16h b;
#pragma unroll
  for (int pp = 0; pp < 8; ++pp) {
    b[2 * pp]     = p[(size_t)(2 * pp) * DHD];
    b[2 * pp + 1] = p[(size_t)(2 * pp + 1) * DHD];
  }
  return b;
}

// ------------------- EMA scan (also emits x as f16) ------------------------
__global__ void ema_kernel(const float* __restrict__ x, float* __restrict__ l2,
                           f16* __restrict__ x16) {
  int c = blockIdx.x * blockDim.x + threadIdx.x;
  if (c >= BB * DD) return;
  int b = c / DD, ch = c % DD;
  const float* xp = x + (size_t)b * TT * DD + ch;
  float* op = l2 + (size_t)b * TT * DD + ch;
  f16* hp = x16 + (size_t)b * TT * DD + ch;
  float s = 0.f;
  for (int ti = 0; ti < TT; ++ti) {
    float xv = xp[(size_t)ti * DD];
    s = 0.9f * s + 0.1f * xv;
    op[(size_t)ti * DD] = s;
    hp[(size_t)ti * DD] = (f16)xv;
  }
}

// ------------------- vectorized fp32 -> f16 conversion ---------------------
__global__ void f32tof16_kernel(const float* __restrict__ src, f16* __restrict__ dst,
                                int nvec4) {   // nvec4 = count/4
  int i = blockIdx.x * blockDim.x + threadIdx.x;
  if (i >= nvec4) return;
  float4 v = ((const float4*)src)[i];
  f16* d = dst + (size_t)i * 4;
  d[0] = (f16)v.x; d[1] = (f16)v.y; d[2] = (f16)v.z; d[3] = (f16)v.w;
}

// ----------------------- WMMA GEMM:  C = A * W^T ---------------------------
// A16: [M,K] f16 row-major, W16: [N,K] f16 row-major.
// Block = 4 waves sharing one 16-row f16 A tile (async-staged to LDS); each
// wave owns a 16x64 output tile. Epilogues: optional bias+SiLU, optional f16
// copy, optional head-split ([b,t,d] -> [b,h,t,dh] fp32 + f16).
__global__ void gemm_wmma_kernel(const f16* __restrict__ A16, const f16* __restrict__ W16,
                                 float* __restrict__ C, f16* __restrict__ C16,
                                 const float* __restrict__ bias,
                                 int M, int N, int K, int silu, int headsplit) {
  __shared__ f16 lds_a[16 * 32];   // 1KB f16 A tile, shared by 4 waves
  int tid  = threadIdx.x;
  int lane = tid & 31;
  int wv   = tid >> 5;
  int tilesN256 = N >> 8;
  int tm = blockIdx.x / tilesN256;
  int tn = (blockIdx.x % tilesN256) * 4 + wv;
  int row0 = tm * 16, col0 = tn * 64;
  if (row0 >= M) return;

  // cooperative A-stage: threads 0..63 copy 16B (8 f16) each
  int ar = tid >> 2, ac = (tid & 3) << 3;

  v8f c0 = {}, c1 = {}, c2 = {}, c3 = {};
  for (int k0 = 0; k0 < K; k0 += 32) {
    const f16* gsrc = A16 + (size_t)(row0 + ar) * K + k0 + ac;
    f16* ldst = &lds_a[ar * 32 + ac];
#if HAVE_ASYNC_LDS
    if (tid < 64)
      __builtin_amdgcn_global_load_async_to_lds_b128(
          (AS1 v4i_vec*)gsrc, (AS3 v4i_vec*)ldst, 0, 0);
    __builtin_amdgcn_s_wait_asynccnt(0);
#else
    if (tid < 64) *(float4*)ldst = *(const float4*)gsrc;
#endif
    __syncthreads();

    if (k0 + 32 < K)
      __builtin_prefetch((const void*)(W16 + (size_t)(col0 + (lane & 15)) * K + k0 + 32), 0, 3);

    v16h a  = load_a_frag_lds16(lds_a, lane);
    v16h b0 = load_b_frag_w16(W16, K, col0,      k0, lane);
    v16h b1 = load_b_frag_w16(W16, K, col0 + 16, k0, lane);
    v16h b2 = load_b_frag_w16(W16, K, col0 + 32, k0, lane);
    v16h b3 = load_b_frag_w16(W16, K, col0 + 48, k0, lane);
    c0 = WMMA16(a, b0, c0);
    c1 = WMMA16(a, b1, c1);
    c2 = WMMA16(a, b2, c2);
    c3 = WMMA16(a, b3, c3);
    __syncthreads();
  }

  int n = lane & 15, half = lane >> 4;
#pragma unroll
  for (int v = 0; v < 8; ++v) {
    int m = row0 + v + 8 * half;
    float vals[4] = { c0[v], c1[v], c2[v], c3[v] };
#pragma unroll
    for (int g = 0; g < 4; ++g) {
      int col = col0 + g * 16 + n;
      float xv = vals[g];
      if (bias) xv += bias[col];
      if (silu) xv = xv / (1.f + __expf(-xv));
      if (headsplit) {
        int bb = m / TT, ti = m % TT;
        int hh = col >> 7, dd = col & 127;
        size_t oidx = (((size_t)bb * HH + hh) * TT + ti) * DHD + dd;
        C[oidx]   = xv;
        C16[oidx] = (f16)xv;
      } else {
        size_t oidx = (size_t)m * N + col;
        if (C)   C[oidx]   = xv;
        if (C16) C16[oidx] = (f16)xv;
      }
    }
  }
}

// ------------------------------ router tail --------------------------------
__global__ void router_kernel(const float* __restrict__ hdn, const float* __restrict__ rw2,
                              const float* __restrict__ rb2, float* __restrict__ lam) {
  int tok = blockIdx.x;
  const float* hp = hdn + (size_t)tok * HDN;
  float p0 = 0.f, p1 = 0.f, p2 = 0.f;
  for (int i = threadIdx.x; i < HDN; i += blockDim.x) {
    float hv = hp[i];
    p0 += hv * rw2[i];
    p1 += hv * rw2[HDN + i];
    p2 += hv * rw2[2 * HDN + i];
  }
#pragma unroll
  for (int off = 16; off; off >>= 1) {
    p0 += __shfl_xor(p0, off, 32);
    p1 += __shfl_xor(p1, off, 32);
    p2 += __shfl_xor(p2, off, 32);
  }
  __shared__ float red[4][3];
  int wv = threadIdx.x >> 5, lane = threadIdx.x & 31;
  if (lane == 0) { red[wv][0] = p0; red[wv][1] = p1; red[wv][2] = p2; }
  __syncthreads();
  if (threadIdx.x == 0) {
    float l0 = red[0][0] + red[1][0] + red[2][0] + red[3][0] + rb2[0];
    float l1 = red[0][1] + red[1][1] + red[2][1] + red[3][1] + rb2[1];
    float l2 = red[0][2] + red[1][2] + red[2][2] + red[3][2] + rb2[2];
    float mx = fmaxf(l0, fmaxf(l1, l2));
    float e0 = __expf(l0 - mx), e1 = __expf(l1 - mx), e2 = __expf(l2 - mx);
    float inv = 1.f / (e0 + e1 + e2);
    lam[(size_t)tok * 3 + 0] = e0 * inv;
    lam[(size_t)tok * 3 + 1] = e1 * inv;
    lam[(size_t)tok * 3 + 2] = e2 * inv;
  }
}

// --------------------- fuse levels (f16 output only) -----------------------
__global__ void fuse_kernel(const float* __restrict__ x, const float* __restrict__ l2,
                            const float* __restrict__ l3mem, const float* __restrict__ lam,
                            f16* __restrict__ fused16) {
  size_t i = (size_t)blockIdx.x * blockDim.x + threadIdx.x;
  size_t tot = (size_t)BB * TT * DD;
  if (i >= tot) return;
  int ch = (int)(i % DD);
  size_t bt = i / DD;
  int b = (int)(bt / TT);
  float la = lam[bt * 3 + 0], lb = lam[bt * 3 + 1], lc = lam[bt * 3 + 2];
  fused16[i] = (f16)(la * x[i] + lb * l2[i] + lc * l3mem[(size_t)b * DD + ch]);
}

// -------------------- flash attention, one wave per q-tile -----------------
// grid: (T/16, H, B), block: 32 threads (single wave). Emits f16 output.
__global__ void attention_kernel(const f16* __restrict__ q16,  // [b,t,d] f16
                                 const f16* __restrict__ kf,   // [b,h,t,dh]
                                 const f16* __restrict__ vf,   // [b,h,t,dh]
                                 f16* __restrict__ out16) {    // [b,t,d] f16
  __shared__ f16 lds_p[16 * 32];
  int lane = threadIdx.x;
  int q0 = blockIdx.x * 16;
  int h  = blockIdx.y;
  int b  = blockIdx.z;

  const float SCALE = 0.08838834764831845f;  // 1/sqrt(128)
  const float NEG_INF = -__builtin_inff();

  const f16* qbh = q16 + (size_t)b * TT * DD + (size_t)h * DHD;  // lda = DD
  const f16* kbh = kf + (((size_t)b * HH + h) * TT) * DHD;
  const f16* vbh = vf + (((size_t)b * HH + h) * TT) * DHD;

  // Q tile 16x128 as 4 A-fragments (raw f16; scale applied post-WMMA)
  v16h qa[4];
#pragma unroll
  for (int i = 0; i < 4; ++i) qa[i] = load_a_frag_g16(qbh, DD, q0, 32 * i, lane);

  v8f o[8];
#pragma unroll
  for (int nt = 0; nt < 8; ++nt) o[nt] = (v8f){};
  float m_run[8], l_run[8];
#pragma unroll
  for (int v = 0; v < 8; ++v) { m_run[v] = NEG_INF; l_run[v] = 0.f; }

  int n = lane & 15, half = lane >> 4;

  for (int j0 = 0; j0 < q0 + 16; j0 += 32) {
    // ---- scores: two 16x16 tiles over a 32-key chunk ----
    v8f s0 = {}, s1 = {};
#pragma unroll
    for (int i = 0; i < 4; ++i) {
      v16h bk0 = load_b_frag_k(kbh, j0,      32 * i, lane);
      v16h bk1 = load_b_frag_k(kbh, j0 + 16, 32 * i, lane);
      s0 = WMMA16(qa[i], bk0, s0);
      s1 = WMMA16(qa[i], bk1, s1);
    }

    // ---- scale + causal mask + online softmax ----
    float alpha[8];
#pragma unroll
    for (int v = 0; v < 8; ++v) {
      int grow = q0 + v + 8 * half;
      float x0 = (j0 + n      <= grow) ? s0[v] * SCALE : NEG_INF;
      float x1 = (j0 + 16 + n <= grow) ? s1[v] * SCALE : NEG_INF;
      float rm = fmaxf(x0, x1);
#pragma unroll
      for (int off = 1; off < 16; off <<= 1) rm = fmaxf(rm, __shfl_xor(rm, off, 32));
      float mnew = fmaxf(m_run[v], rm);
      alpha[v] = __expf(m_run[v] - mnew);
      float e0 = __expf(x0 - mnew);
      float e1 = __expf(x1 - mnew);
      float rs = e0 + e1;
#pragma unroll
      for (int off = 1; off < 16; off <<= 1) rs += __shfl_xor(rs, off, 32);
      l_run[v] = l_run[v] * alpha[v] + rs;
      m_run[v] = mnew;
      lds_p[(v + 8 * half) * 32 + n]      = (f16)e0;
      lds_p[(v + 8 * half) * 32 + 16 + n] = (f16)e1;
    }
    __syncthreads();   // single-wave WG: orders ds_store -> ds_load

#pragma unroll
    for (int nt = 0; nt < 8; ++nt)
#pragma unroll
      for (int v = 0; v < 8; ++v) o[nt][v] *= alpha[v];

    // ---- P (16x32) from LDS in A-layout ----
    v16h pa = load_a_frag_lds16(lds_p, lane);

    // ---- O += P @ V over dh=128 (8 n-tiles) ----
#pragma unroll
    for (int nt = 0; nt < 8; ++nt) {
      v16h bv = load_b_frag_v(vbh, j0, nt * 16, lane);
      o[nt] = WMMA16(pa, bv, o[nt]);
    }
    __syncthreads();   // protect lds_p before next chunk
  }

  // ---- normalize & store (f16) ----
#pragma unroll
  for (int v = 0; v < 8; ++v) {
    float inv = 1.f / l_run[v];
    int grow = q0 + v + 8 * half;
    f16* orow = out16 + ((size_t)b * TT + grow) * DD + (size_t)h * DHD;
#pragma unroll
    for (int nt = 0; nt < 8; ++nt) orow[nt * 16 + n] = (f16)(o[nt][v] * inv);
  }
}

// ------------------------------- launcher ----------------------------------
extern "C" void kernel_launch(void* const* d_in, const int* in_sizes, int n_in,
                              void* d_out, int out_size, void* d_ws, size_t ws_size,
                              hipStream_t stream) {
  const float* x      = (const float*)d_in[0];
  const float* l3mem  = (const float*)d_in[1];
  const float* wq     = (const float*)d_in[2];
  const float* wk     = (const float*)d_in[3];
  const float* wv     = (const float*)d_in[4];
  const float* wo     = (const float*)d_in[5];
  const float* rw1    = (const float*)d_in[6];
  const float* rb1    = (const float*)d_in[7];
  const float* rw2    = (const float*)d_in[8];
  const float* rb2    = (const float*)d_in[9];

  const size_t BTD = (size_t)BB * TT * DD;          // 4,194,304
  float* out_main = (float*)d_out;                  // [b,t,d] fp32
  float* out_kh   = out_main + BTD;                 // [b,h,t,dh] fp32
  float* out_vh   = out_kh + BTD;                   // [b,h,t,dh] fp32
  float* out_lam  = out_vh + BTD;                   // [b,t,3]

  // workspace: fp32 region then f16 region (~108 MB total)
  float* ws   = (float*)d_ws;
  float* l2   = ws;                                 // BTD  fp32
  float* hdn  = l2 + BTD;                           // BTD/2 fp32
  f16* h0     = (f16*)(hdn + BTD / 2);
  f16* x16    = h0;                                 // BTD
  f16* q16    = x16 + BTD;                          // BTD
  f16* fus16  = q16 + BTD;                          // BTD
  f16* kh16   = fus16 + BTD;                        // BTD
  f16* vh16   = kh16 + BTD;                         // BTD
  f16* ao16   = vh16 + BTD;                         // BTD (attention out)
  f16* wq16   = ao16 + BTD;                         // DD*DD
  f16* wk16   = wq16 + (size_t)DD * DD;
  f16* wv16   = wk16 + (size_t)DD * DD;
  f16* wo16   = wv16 + (size_t)DD * DD;
  f16* rw116  = wo16 + (size_t)DD * DD;             // HDN*DD

  const int M = BB * TT;   // 2048
  dim3 blk128(128), blk256(256), blk32(32);

  // 1) EMA scan (l2) + x -> f16
  ema_kernel<<<(BB * DD + 255) / 256, blk256, 0, stream>>>(x, l2, x16);

  // 1b) weights -> f16 (vectorized, 4 elems/thread)
  int wv4 = (DD * DD) / 4, rv4 = (HDN * DD) / 4;
  f32tof16_kernel<<<(wv4 + 255) / 256, blk256, 0, stream>>>(wq, wq16, wv4);
  f32tof16_kernel<<<(wv4 + 255) / 256, blk256, 0, stream>>>(wk, wk16, wv4);
  f32tof16_kernel<<<(wv4 + 255) / 256, blk256, 0, stream>>>(wv, wv16, wv4);
  f32tof16_kernel<<<(wv4 + 255) / 256, blk256, 0, stream>>>(wo, wo16, wv4);
  f32tof16_kernel<<<(rv4 + 255) / 256, blk256, 0, stream>>>(rw1, rw116, rv4);

  // 2) q = x @ wq^T (f16 out only) : blocks = (M/16)*(N/256) = 1024
  gemm_wmma_kernel<<<1024, blk128, 0, stream>>>(x16, wq16, nullptr, q16, nullptr,
                                                M, DD, DD, 0, 0);

  // 3) hdn = silu(q @ rw1^T + rb1) : blocks = 128*4 = 512
  gemm_wmma_kernel<<<512, blk128, 0, stream>>>(q16, rw116, hdn, nullptr, rb1,
                                               M, HDN, DD, 1, 0);

  // 4) lam = softmax(hdn @ rw2^T + rb2)
  router_kernel<<<M, blk128, 0, stream>>>(hdn, rw2, rb2, out_lam);

  // 5) fused = lam0*x + lam1*l2 + lam2*l3  (f16)
  fuse_kernel<<<(unsigned)((BTD + 255) / 256), blk256, 0, stream>>>(x, l2, l3mem,
                                                                   out_lam, fus16);

  // 6) K/V projections with fused head-split epilogue (fp32 d_out + f16 cache)
  gemm_wmma_kernel<<<1024, blk128, 0, stream>>>(fus16, wk16, out_kh, kh16, nullptr,
                                                M, DD, DD, 0, 1);
  gemm_wmma_kernel<<<1024, blk128, 0, stream>>>(fus16, wv16, out_vh, vh16, nullptr,
                                                M, DD, DD, 0, 1);

  // 7) causal attention (f16 out)
  dim3 agrid(TT / 16, HH, BB);
  attention_kernel<<<agrid, blk32, 0, stream>>>(q16, kh16, vh16, ao16);

  // 8) out = attnout @ wo^T (fp32 out)
  gemm_wmma_kernel<<<1024, blk128, 0, stream>>>(ao16, wo16, out_main, nullptr, nullptr,
                                                M, DD, DD, 0, 0);
}
